// LSTM_29746943492607
// MI455X (gfx1250) — matrix operand order repaired and checked by reading, not compiled, not run on hardware
//
#include <hip/hip_runtime.h>

// Problem constants (match reference)
#define Cc 3
#define Tt 10
#define Dd 5
#define Hh 8           // hidden per direction
#define GATES 32       // 4*H
#define FEAT0 15       // C*D (layer-0 input features)
#define FIN 16         // padded per-layer input feature count (2*H, padded 15)
#define WPB 4          // waves per block (wave32)
#define THREADS (WPB * 32)

typedef _Float16 h16;
typedef __attribute__((ext_vector_type(16))) _Float16 v16h;
typedef __attribute__((ext_vector_type(2)))  __fp16   v2fp;
typedef __attribute__((ext_vector_type(8)))  float    v8f;

union AFrag { v16h v; h16 h[16]; uint4 q[2]; };
union BFrag { v16h v; h16 h[16]; };
union Pk    { v2fp p; int i; h16 h[2]; };

// ---- fast transcendentals: v_exp_f32 / v_rcp_f32, no IEEE div chains ------
__device__ __forceinline__ float fast_exp(float x) {
#if __has_builtin(__builtin_amdgcn_exp2f)
    return __builtin_amdgcn_exp2f(x * 1.4426950408889634f);
#else
    return __expf(x);
#endif
}
__device__ __forceinline__ float fast_rcp(float x) {
#if __has_builtin(__builtin_amdgcn_rcpf)
    return __builtin_amdgcn_rcpf(x);
#else
    return 1.0f / x;
#endif
}
__device__ __forceinline__ float fast_sigmoid(float x) {
    return fast_rcp(1.0f + fast_exp(-x));
}
// xor-8 lane swap via ds_swizzle (group-of-32: and=0x1f, or=0, xor=0x08)
__device__ __forceinline__ int swap8i(int v) {
    return __builtin_amdgcn_ds_swizzle(v, 0x201F);
}

// 3-layer bidirectional LSTM + output projection. One wave32 = 16-batch tile.
// Per layer both directions are interleaved: 4x v_wmma_f32_16x16x32_f16 per
// step (2 gate N-tiles x 2 independent recurrences) on fused [Wih|Whh|0] K=32
// weights; bias rides in the C operand. f16 inputs, f32 accumulate. The h
// recurrence round-trips through the next-layer activation buffer directly.
__global__ __launch_bounds__(THREADS)
void lstm_bi3_wmma_kernel(const float* __restrict__ x,
                          const float* __restrict__ Wih0,
                          const float* __restrict__ WihR,
                          const float* __restrict__ Whh,
                          const float* __restrict__ bvec,
                          const float* __restrict__ Wout,
                          const float* __restrict__ bout,
                          float* __restrict__ out,
                          long Btot)
{
    __shared__ h16   s_W[6 * 32 * 32];          // fused [l*2+dir][K=32][N=32]
    __shared__ float s_bias[6 * 32];
    __shared__ float s_wout[Tt * FIN];
    __shared__ __align__(16) h16 s_hin[WPB][2][16][Tt][FIN]; // act double-buffer

    const int tid  = threadIdx.x;
    const int wave = tid >> 5;
    const int lane = tid & 31;
    const int n    = lane & 15;   // tile column (N) / A-matrix M row
    const int grp  = lane >> 4;   // 0: C rows M=0..7, A K-low ; 1: M=8..15, K-high

    // ---------------- stage weights (block-cooperative, f32 -> f16) ---------
    for (int idx = tid; idx < 6 * 32 * 32; idx += THREADS) {
        const int combo = idx >> 10;        // l*2 + dir
        const int K = (idx >> 5) & 31;
        const int N = idx & 31;
        const int l = combo >> 1, dir = combo & 1;
        const int kin = (l == 0) ? FEAT0 : 2 * Hh;
        float v = 0.0f;
        if (K < kin) {
            v = (l == 0) ? Wih0[(dir * GATES + N) * FEAT0 + K]
                         : WihR[(((l - 1) * 2 + dir) * GATES + N) * (2 * Hh) + K];
        } else if (K >= 16 && K < 16 + Hh) {
            v = Whh[((l * 2 + dir) * GATES + N) * Hh + (K - 16)];
        }
        s_W[idx] = (h16)v;
    }
    for (int idx = tid; idx < 6 * 32; idx += THREADS)   s_bias[idx] = bvec[idx];
    for (int idx = tid; idx < Tt * FIN; idx += THREADS) s_wout[idx] = Wout[idx];

    // ---------------- stage x tile into layer-0 input buffer (per-wave) -----
    const long tile = (long)blockIdx.x * WPB + wave;
    const long b0   = tile * 16;
    for (int idx = lane; idx < 16 * Tt * FIN; idx += 32) {
        const int f = idx & 15;
        const int t = (idx >> 4) % Tt;
        const int M = idx / (Tt * FIN);
        float v = 0.0f;
        if (f < FEAT0) {
            long bb = b0 + M; if (bb >= Btot) bb = Btot - 1;
            const int c = f / Dd, d = f % Dd;
            v = x[(bb * Cc + c) * (Tt * Dd) + t * Dd + d];
        }
        s_hin[wave][0][M][t][f] = (h16)v;
    }
    __syncthreads();

    // tile-1 activation: n<8 -> tanh(g) = 2*sig(2x)-1 ; n>=8 -> sig(o)
    const float kin1 = (n < 8) ? 2.0f : 1.0f;
    const float kof1 = (n < 8) ? -1.0f : 0.0f;
    const uint4 zq = make_uint4(0, 0, 0, 0);

    // ---------------- 3 layers, both directions interleaved -----------------
    int cur = 0;
    for (int l = 0; l < 3; ++l) {
        const int nxt = cur ^ 1;
        // B fragments for both dirs and both gate tiles (loop-invariant)
        BFrag bf[2][2];
        float bias[2][2];
        #pragma unroll
        for (int dir = 0; dir < 2; ++dir) {
            const int combo = l * 2 + dir;
            #pragma unroll
            for (int hh = 0; hh < 16; ++hh) {
                const int K = (hh & 7) + 8 * grp + 16 * (hh >> 3);
                bf[dir][0].h[hh] = s_W[(combo << 10) + (K << 5) + n];
                bf[dir][1].h[hh] = s_W[(combo << 10) + (K << 5) + 16 + n];
            }
            bias[dir][0] = s_bias[combo * 32 + n];
            bias[dir][1] = s_bias[combo * 32 + 16 + n];
        }
        v8f cb[2][2];   // hoisted bias accumulators (column-constant C operand)
        #pragma unroll
        for (int dir = 0; dir < 2; ++dir)
            #pragma unroll
            for (int tt = 0; tt < 2; ++tt)
                #pragma unroll
                for (int r = 0; r < 8; ++r) cb[dir][tt][r] = bias[dir][tt];

        float cst[2][8];
        #pragma unroll
        for (int dir = 0; dir < 2; ++dir)
            #pragma unroll
            for (int r = 0; r < 8; ++r) cst[dir][r] = 0.0f;
        asm volatile("" ::: "memory");

        for (int s = 0; s < Tt; ++s) {
            const int tf = s, tb = Tt - 1 - s;      // fwd / bwd timestep
            AFrag a[2];
            a[0].q[0] = *(const uint4*)&s_hin[wave][cur][n][tf][grp * 8];
            a[1].q[0] = *(const uint4*)&s_hin[wave][cur][n][tb][grp * 8];
            if (s == 0) {                           // wave-uniform: h0 = 0
                a[0].q[1] = zq;
                a[1].q[1] = zq;
            } else {                                // h_prev lives in nxt buffer
                uint4 hf = *(const uint4*)&s_hin[wave][nxt][n][tf - 1][0];
                uint4 hb = *(const uint4*)&s_hin[wave][nxt][n][tb + 1][Hh];
                a[0].q[1] = grp ? zq : hf;          // grp1 K=24..31 is zero pad
                a[1].q[1] = grp ? zq : hb;
            }

            v8f g0f = __builtin_amdgcn_wmma_f32_16x16x32_f16(
                        false, a[0].v, false, bf[0][0].v, (short)0, cb[0][0], false, false);
            v8f g1f = __builtin_amdgcn_wmma_f32_16x16x32_f16(
                        false, a[0].v, false, bf[0][1].v, (short)0, cb[0][1], false, false);
            v8f g0b = __builtin_amdgcn_wmma_f32_16x16x32_f16(
                        false, a[1].v, false, bf[1][0].v, (short)0, cb[1][0], false, false);
            v8f g1b = __builtin_amdgcn_wmma_f32_16x16x32_f16(
                        false, a[1].v, false, bf[1][1].v, (short)0, cb[1][1], false, false);

            // D layout: lane n holds col n, VGPR r holds row M = 8*grp + r.
            // Activate first (i,f both sigmoid; tile1 via per-lane constants),
            // pack the pair to f16x2 and do ONE xor-8 swap for (f,o).
            h16 hnf[8], hnb[8];
            #pragma unroll
            for (int r = 0; r < 8; ++r) {
                // forward
                float a0 = fast_sigmoid(g0f[r]);                      // sig(i)|sig(f)
                float a1 = fast_sigmoid(kin1 * g1f[r]) * kin1 + kof1; // tanh(g)|sig(o)
                Pk pf; pf.p = __builtin_amdgcn_cvt_pkrtz(a0, a1);
                Pk sf; sf.i = swap8i(pf.i);
                float fr  = (float)sf.h[0];                           // sig(f)
                float orr = (float)sf.h[1];                           // sig(o)
                float cc = fr * cst[0][r] + a0 * a1;
                cst[0][r] = cc;
                hnf[r] = (h16)(orr * (2.0f * fast_sigmoid(2.0f * cc) - 1.0f));
                // backward
                float b0a = fast_sigmoid(g0b[r]);
                float b1a = fast_sigmoid(kin1 * g1b[r]) * kin1 + kof1;
                Pk pb; pb.p = __builtin_amdgcn_cvt_pkrtz(b0a, b1a);
                Pk sb; sb.i = swap8i(pb.i);
                float fb = (float)sb.h[0];
                float ob = (float)sb.h[1];
                float cc2 = fb * cst[1][r] + b0a * b1a;
                cst[1][r] = cc2;
                hnb[r] = (h16)(ob * (2.0f * fast_sigmoid(2.0f * cc2) - 1.0f));
            }
            if (n < 8) {    // lanes n<8 own (M = 8*grp + r, j = n)
                #pragma unroll
                for (int r = 0; r < 8; ++r) {
                    const int M = grp * 8 + r;
                    s_hin[wave][nxt][M][tf][n]      = hnf[r];
                    s_hin[wave][nxt][M][tb][Hh + n] = hnb[r];
                }
            }
            asm volatile("" ::: "memory");   // keep DS order for next step
        }
        cur ^= 1;
    }

    // ---------------- output projection: out[b] = flat . Wout + bout --------
    asm volatile("" ::: "memory");
    if (lane < 16 && (b0 + lane) < Btot) {
        float acc = bout[0];
        #pragma unroll
        for (int t = 0; t < Tt; ++t) {
            #pragma unroll
            for (int j = 0; j < FIN; ++j)
                acc += (float)s_hin[wave][cur][lane][t][j] * s_wout[t * FIN + j];
        }
        out[b0 + lane] = acc;
    }
}

extern "C" void kernel_launch(void* const* d_in, const int* in_sizes, int n_in,
                              void* d_out, int out_size, void* d_ws, size_t ws_size,
                              hipStream_t stream) {
    const float* x    = (const float*)d_in[0];
    const float* Wih0 = (const float*)d_in[1];
    const float* WihR = (const float*)d_in[2];
    const float* Whh  = (const float*)d_in[3];
    const float* bb   = (const float*)d_in[4];
    const float* Wout = (const float*)d_in[5];
    const float* bout = (const float*)d_in[6];
    float* out = (float*)d_out;

    const long Btot = (long)in_sizes[0] / (Cc * Tt * Dd);   // 262144
    const int  nblk = (int)((Btot + 16 * WPB - 1) / (16 * WPB));
    hipLaunchKernelGGL(lstm_bi3_wmma_kernel, dim3(nblk), dim3(THREADS), 0, stream,
                       x, Wih0, WihR, Whh, bb, Wout, bout, out, Btot);
}
